// GCNModel_36541581754800
// MI455X (gfx1250) — matrix operand order repaired and checked by reading, not compiled, not run on hardware
//
#include <hip/hip_runtime.h>
#include <hip/hip_bf16.h>
#include <math.h>

// ---------------------------------------------------------------------------
// GCN (2x GCNConv + global_max_pool + MLP head) for MI455X / gfx1250.
//
// Dense transforms: WMMA pipe (v_wmma_f32_16x16x32_bf16, bf16 in, f32 accum).
// Fusions to cut HBM traffic (the real bottleneck at 23.3 TB/s):
//   - GEMM epilogue writes both m and agg = m*dinv^2 (self-loop init).
//   - relu(agg+b1) fused into GEMM-2's A-tile staging (f32->bf16 convert).
//   - relu(agg+b2) fused into the pooling kernel before the bit-atomicMax.
// Edge propagation stays a coalesced gather + f32 global-atomic scatter.
// ---------------------------------------------------------------------------

typedef __bf16 bf16_t;
typedef bf16_t bf16x16 __attribute__((ext_vector_type(16)));
typedef float  floatx8 __attribute__((ext_vector_type(8)));

#define FDIM 128            // feature / hidden width (fixed by reference)
#define GEMM_ROWS 64        // rows per block (4 waves x 16 rows)
#define GEMM_THREADS 128    // 4 wave32
#define LDS_STRIDE 132      // 132 bf16 = 264B = 66 dwords -> conflict-free rows

// ---------------------------------------------------------------------------
// degree / normalization helpers
// ---------------------------------------------------------------------------
__global__ void k_fill_f32(float* p, int n, float v) {
    int i = blockIdx.x * blockDim.x + threadIdx.x;
    if (i < n) p[i] = v;
}

__global__ void k_degree(float* deg, const int* dst, int E) {
    int i = blockIdx.x * blockDim.x + threadIdx.x;
    if (i < E) atomicAdd(&deg[dst[i]], 1.0f);
}

__global__ void k_rsqrt_inplace(float* d, int n) {
    int i = blockIdx.x * blockDim.x + threadIdx.x;
    if (i < n) {
        float v = d[i];
        d[i] = (v > 0.0f) ? rsqrtf(v) : 0.0f;
    }
}

// ---------------------------------------------------------------------------
// WMMA GEMM: C[nrows x 128] = act(A)[nrows x 128] * W[128 x 128]
//   act(A) = relu(A + abias) when abias != nullptr (fused prev-layer epilogue)
//   agg[r,c] = C[r,c] * dinv[r]^2 when agg != nullptr (fused self-loop init)
// Per block: 64 rows, 4 waves; each wave owns a 16x128 strip (8 accumulators),
// A-fragment reused across the 8 column tiles, K-loop of 4 x (K=32).
// In-place agg==A-region is safe: block reads exactly the rows it writes, and
// all global reads of the tile complete (into LDS) before epilogue stores.
// ---------------------------------------------------------------------------
__global__ __launch_bounds__(GEMM_THREADS)
void k_gemm_wmma(const float* __restrict__ A, const float* __restrict__ W,
                 const float* __restrict__ abias,
                 float* __restrict__ C, float* __restrict__ agg,
                 const float* __restrict__ dinv, int nrows) {
    __shared__ bf16_t Alds[GEMM_ROWS][LDS_STRIDE];   // [row][k]
    __shared__ bf16_t Wlds[FDIM][LDS_STRIDE];        // transposed: [col][k]

    const int tid = threadIdx.x;
    const int rowBase = blockIdx.x * GEMM_ROWS;

    // --- stage A tile (64 x 128 f32 -> bf16), coalesced float4 loads,
    //     optional fused bias+relu from the previous layer ---
    for (int idx = tid; idx < GEMM_ROWS * (FDIM / 4); idx += GEMM_THREADS) {
        int r  = idx >> 5;          // /32 float4 per row
        int c4 = idx & 31;
        int grow = rowBase + r;
        float4 v = make_float4(0.f, 0.f, 0.f, 0.f);
        if (grow < nrows) v = ((const float4*)A)[(size_t)grow * (FDIM / 4) + c4];
        int c = c4 * 4;
        if (abias) {
            v.x = fmaxf(v.x + abias[c + 0], 0.0f);
            v.y = fmaxf(v.y + abias[c + 1], 0.0f);
            v.z = fmaxf(v.z + abias[c + 2], 0.0f);
            v.w = fmaxf(v.w + abias[c + 3], 0.0f);
        }
        Alds[r][c + 0] = (bf16_t)v.x;
        Alds[r][c + 1] = (bf16_t)v.y;
        Alds[r][c + 2] = (bf16_t)v.z;
        Alds[r][c + 3] = (bf16_t)v.w;
    }
    // --- stage W transposed (128 x 128 f32 -> bf16) ---
    for (int idx = tid; idx < FDIM * (FDIM / 4); idx += GEMM_THREADS) {
        int k  = idx >> 5;
        int c4 = idx & 31;
        float4 v = ((const float4*)W)[(size_t)k * (FDIM / 4) + c4];
        int c = c4 * 4;
        Wlds[c + 0][k] = (bf16_t)v.x;
        Wlds[c + 1][k] = (bf16_t)v.y;
        Wlds[c + 2][k] = (bf16_t)v.z;
        Wlds[c + 3][k] = (bf16_t)v.w;
    }
    __syncthreads();

    const int lane = tid & 31;
    const int wv   = tid >> 5;
    const int m16  = lane & 15;
    const int hi   = (lane >> 4) & 1;
    const int mrow = wv * 16 + m16;      // local row in Alds

    union Frag { bf16x16 v; bf16_t s[16]; };
    union Acc  { floatx8 v; float  s[8];  };
    Acc acc[8];
#pragma unroll
    for (int t = 0; t < 8; ++t)
#pragma unroll
        for (int r = 0; r < 8; ++r) acc[t].s[r] = 0.0f;

#pragma unroll
    for (int kk = 0; kk < FDIM; kk += 32) {
        // A fragment: 16x32 bf16, CDNA5 layout:
        //   lanes 0-15 hold K = {0..7, 16..23}, lanes 16-31 hold K = {8..15, 24..31}
        Frag af;
#pragma unroll
        for (int e = 0; e < 16; ++e) {
            int K = kk + ((e < 8) ? e : (e + 8)) + hi * 8;
            af.s[e] = Alds[mrow][K];
        }
#pragma unroll
        for (int t = 0; t < 8; ++t) {
            // B fragment: 32x16 bf16, lane = column; lanes 0-15: K={0..7, 16..23},
            // lanes 16-31: K={8..15, 24..31} (2 bf16 per VGPR).
            Frag bf;
            int col = t * 16 + m16;
#pragma unroll
            for (int e = 0; e < 16; ++e) {
                int K = kk + ((e < 8) ? e : (e + 8)) + hi * 8;
                bf.s[e] = Wlds[col][K];
            }
            acc[t].v = __builtin_amdgcn_wmma_f32_16x16x32_bf16(
                false, af.v, false, bf.v, (short)0, acc[t].v, false, false);
        }
    }

    // --- epilogue: C/D layout: VGPR r -> M = r (+8 for hi lanes), N = lane&15.
    //     Store m and (optionally) the self-loop-initialized aggregate. ---
#pragma unroll
    for (int t = 0; t < 8; ++t) {
#pragma unroll
        for (int r = 0; r < 8; ++r) {
            int grow = rowBase + wv * 16 + r + hi * 8;
            int gcol = t * 16 + m16;
            if (grow < nrows) {
                float val = acc[t].s[r];
                C[(size_t)grow * FDIM + gcol] = val;
                if (agg) {
                    float w = dinv[grow];
                    agg[(size_t)grow * FDIM + gcol] = val * w * w;
                }
            }
        }
    }
}

// ---------------------------------------------------------------------------
// edge scatter: agg[dst] += m[src] * dinv[src]*dinv[dst]
// 128 lanes per edge -> coalesced feature-row gather and atomic scatter.
// ---------------------------------------------------------------------------
__global__ void k_edge_scatter(float* __restrict__ agg, const float* __restrict__ m,
                               const float* __restrict__ dinv,
                               const int* __restrict__ src, const int* __restrict__ dst,
                               int total /* E*128 */) {
    int gid = blockIdx.x * blockDim.x + threadIdx.x;
    if (gid < total) {
        int e = gid >> 7;
        int f = gid & 127;
        int s = src[e];
        int d = dst[e];
        float w = dinv[s] * dinv[d];
        atomicAdd(&agg[(size_t)d * FDIM + f], m[(size_t)s * FDIM + f] * w);
    }
}

// ---------------------------------------------------------------------------
// global max pool with fused bias+relu of layer 2.
// relu output >= 0, so float-bit atomicMax is monotone vs. 0-initialized pool.
// ---------------------------------------------------------------------------
__global__ void k_zero_u32(unsigned* p, int n) {
    int i = blockIdx.x * blockDim.x + threadIdx.x;
    if (i < n) p[i] = 0u;
}

__global__ void k_pool_bias_relu_max(unsigned* __restrict__ gpool,
                                     const float* __restrict__ agg,
                                     const float* __restrict__ b,
                                     const int* __restrict__ batch, int total) {
    int gid = blockIdx.x * blockDim.x + threadIdx.x;
    if (gid < total) {
        int node = gid >> 7;
        int f = gid & 127;
        float v = fmaxf(agg[gid] + b[f], 0.0f);   // h2 element, >= 0
        atomicMax(&gpool[(size_t)batch[node] * FDIM + f], __float_as_uint(v));
    }
}

// ---------------------------------------------------------------------------
// head: z = relu(g@W3+b3); logits = z@W4+b4; log_softmax. G=64 -> one block.
// ---------------------------------------------------------------------------
__global__ __launch_bounds__(256)
void k_head(const unsigned* __restrict__ gbits, const float* __restrict__ W3,
            const float* __restrict__ b3, const float* __restrict__ W4,
            const float* __restrict__ b4, float* __restrict__ out, int G) {
    __shared__ float zs[64 * FDIM];
    int tid = threadIdx.x;
    for (int o = tid; o < G * FDIM; o += 256) {
        int row = o >> 7;
        int col = o & 127;
        float s = b3[col];
#pragma unroll 4
        for (int k = 0; k < FDIM; ++k)
            s += __uint_as_float(gbits[(size_t)row * FDIM + k]) * W3[(size_t)k * FDIM + col];
        zs[o] = fmaxf(s, 0.0f);
    }
    __syncthreads();
    if (tid < G) {
        float l0 = b4[0], l1 = b4[1];
#pragma unroll 4
        for (int k = 0; k < FDIM; ++k) {
            float z = zs[tid * FDIM + k];
            l0 += z * W4[k * 2 + 0];
            l1 += z * W4[k * 2 + 1];
        }
        float mx  = fmaxf(l0, l1);
        float lse = mx + logf(expf(l0 - mx) + expf(l1 - mx));
        out[tid * 2 + 0] = l0 - lse;
        out[tid * 2 + 1] = l1 - lse;
    }
}

// ---------------------------------------------------------------------------
// launcher
// ---------------------------------------------------------------------------
extern "C" void kernel_launch(void* const* d_in, const int* in_sizes, int n_in,
                              void* d_out, int out_size, void* d_ws, size_t ws_size,
                              hipStream_t stream) {
    const float* x  = (const float*)d_in[0];
    const float* W1 = (const float*)d_in[1];
    const float* b1 = (const float*)d_in[2];
    const float* W2 = (const float*)d_in[3];
    const float* b2 = (const float*)d_in[4];
    const float* W3 = (const float*)d_in[5];
    const float* b3 = (const float*)d_in[6];
    const float* W4 = (const float*)d_in[7];
    const float* b4 = (const float*)d_in[8];
    const int* edge_index = (const int*)d_in[9];
    const int* batch      = (const int*)d_in[10];

    const int N = in_sizes[0] / FDIM;
    const int E = in_sizes[9] / 2;
    const int G = 64;  // num_graphs (fixed by reference setup)

    const int* src = edge_index;       // edge_index[0]
    const int* dst = edge_index + E;   // edge_index[1]

    // workspace layout (floats): dinv[Npad] | bufA[N*128] | bufB[N*128] | gpool[G*128]
    float* ws    = (float*)d_ws;
    size_t Npad  = ((size_t)N + 63) & ~(size_t)63;
    float* dinv  = ws;
    float* bufA  = ws + Npad;
    float* bufB  = bufA + (size_t)N * FDIM;
    float* gpool = bufB + (size_t)N * FDIM;

    const int nodeElems = N * FDIM;      // 6.4M
    const int edgeElems = E * FDIM;      // 81.9M, fits int32

    dim3 b256(256);
    int gN    = (N + 255) / 256;
    int gE    = (E + 255) / 256;
    int gNF   = (nodeElems + 255) / 256;
    int gEF   = (edgeElems + 255) / 256;
    int gGemm = (N + GEMM_ROWS - 1) / GEMM_ROWS;

    // ---- degree normalization: deg = 1 (self loop) + scatter count; dinv = rsqrt ----
    k_fill_f32<<<gN, b256, 0, stream>>>(dinv, N, 1.0f);
    k_degree<<<gE, b256, 0, stream>>>(dinv, dst, E);
    k_rsqrt_inplace<<<gN, b256, 0, stream>>>(dinv, N);

    // ---- layer 1: m1 = x@W1 -> bufA, agg1 = m1*dinv^2 -> bufB (fused),
    //      then scatter edges into agg1 ----
    k_gemm_wmma<<<gGemm, GEMM_THREADS, 0, stream>>>(x, W1, nullptr,
                                                    bufA, bufB, dinv, N);
    k_edge_scatter<<<gEF, b256, 0, stream>>>(bufB, bufA, dinv, src, dst, edgeElems);

    // ---- layer 2: m2 = relu(agg1+b1)@W2 -> bufA (bias+relu fused in staging),
    //      agg2 = m2*dinv^2 -> bufB (in-place over agg1: per-block exclusive),
    //      then scatter edges into agg2 ----
    k_gemm_wmma<<<gGemm, GEMM_THREADS, 0, stream>>>(bufB, W2, b1,
                                                    bufA, bufB, dinv, N);
    k_edge_scatter<<<gEF, b256, 0, stream>>>(bufB, bufA, dinv, src, dst, edgeElems);

    // ---- global max pool with fused relu(agg2 + b2) ----
    int gG = (G * FDIM + 255) / 256;
    k_zero_u32<<<gG, b256, 0, stream>>>((unsigned*)gpool, G * FDIM);
    k_pool_bias_relu_max<<<gNF, b256, 0, stream>>>((unsigned*)gpool, bufB, b2,
                                                   batch, nodeElems);

    // ---- MLP head + log_softmax -> d_out [G x 2] ----
    k_head<<<1, 256, 0, stream>>>((const unsigned*)gpool, W3, b3, W4, b4,
                                  (float*)d_out, G);
}
